// MambaBlock_33749853012608
// MI455X (gfx1250) — matrix operand order repaired and checked by reading, not compiled, not run on hardware
//
#include <hip/hip_runtime.h>

// ---------------------------------------------------------------------------
// Mamba block forward for MI455X (gfx1250, wave32, WMMA + TDM).
//   GEMM1/GEMM2 -> tensor_load_to_lds double-buffered tiles in LDS,
//                  ds_load_b128 fragments, v_wmma_f32_16x16x32_bf16
//   scan        -> fp32 VALU, 1 thread per channel, 16 states in registers
// ---------------------------------------------------------------------------

#define D_MODEL 1024
#define D_STATE 16
#define D_CONV  4
#define D_INNER 2048
#define TLEN    2048
#define NXZ     (2 * D_INNER)   // 4096

// LDS tile: 64 rows x 64 bf16, padded +8 bf16 per row by TDM (pad feature)
#define TILE_K   64
#define LDS_PITCH 72            // 64 + 4 DWORDs padding = 72 bf16 elements

typedef __attribute__((ext_vector_type(16))) __bf16 v16bf;
typedef __attribute__((ext_vector_type(8)))  __bf16 v8bf;
typedef __attribute__((ext_vector_type(8)))  float  v8f;
typedef __attribute__((ext_vector_type(4)))  unsigned int v4u;
typedef __attribute__((ext_vector_type(8)))  int v8i_;
typedef __attribute__((ext_vector_type(4)))  int v4i_;

static __device__ __forceinline__ __bf16 f2bf(float f) {
  // round-to-nearest-even fp32 -> bf16
  unsigned u = __builtin_bit_cast(unsigned, f);
  unsigned r = (u + 0x7FFFu + ((u >> 16) & 1u)) >> 16;
  unsigned short s = (unsigned short)r;
  return __builtin_bit_cast(__bf16, s);
}

static __device__ __forceinline__ unsigned lds_off(const void* p) {
  // generic pointer to LDS: low 32 bits are the wave-relative LDS byte offset
  return (unsigned)(uintptr_t)p;
}

static __device__ __forceinline__ v16bf ld_frag(const __bf16* p0, const __bf16* p1) {
  v8bf a0 = *(const v8bf*)p0;
  v8bf a1 = *(const v8bf*)p1;
  return __builtin_shufflevector(a0, a1,
      0,1,2,3,4,5,6,7,8,9,10,11,12,13,14,15);
}

// ---------------------------------------------------------------------------
// TDM: DMA a 64x64 bf16 tile (row stride = stride0 elements) from global
// memory into LDS, adding 4 DWORDs of padding every 32 DWORDs (one row),
// i.e. LDS row pitch = 72 bf16. D# layout per CDNA5 ISA.
// ---------------------------------------------------------------------------
static __device__ __forceinline__ void tdm_load_tile64(
    const __bf16* gptr, unsigned lds_byte_off,
    unsigned rem_d0, unsigned rem_d1, unsigned stride0) {
  unsigned long long ga = (unsigned long long)(uintptr_t)gptr;
  v4u g0 = {
      1u,                                   // count=1, user mode, no gather
      lds_byte_off,                         // lds_addr
      (unsigned)ga,                         // global_addr[31:0]
      ((unsigned)(ga >> 32) & 0x01FFFFFFu)  // global_addr[56:32]
          | 0x80000000u                     // type=2 ("image")
  };
  v8i_ g1 = {
      (int)((1u << 16)        // data_size = 2 bytes
            | (1u << 20)      // pad_enable
            | (4u << 22)      // pad_interval code 4 -> every 32 DWORDs
            | (3u << 25)),    // pad_amount  code 3 -> 4 DWORDs (8 bf16)
      (int)((rem_d0 & 0xFFFFu) << 16),                              // dim0 lo
      (int)(((rem_d0 >> 16) & 0xFFFFu) | ((rem_d1 & 0xFFFFu) << 16)),
      (int)(((rem_d1 >> 16) & 0xFFFFu) | (64u << 16)),              // tile_dim0=64
      (int)64,                                                      // tile_dim1=64, tile_dim2=0
      (int)stride0,                                                 // dim0_stride[31:0]
      0, 0
  };
  v4i_ zero4 = {0, 0, 0, 0};
  v8i_ zero8 = {0, 0, 0, 0, 0, 0, 0, 0};
  // 6-arg form (amdgpu-toolchain): (g0, g1, g2, g3, extra, cpol)
  __builtin_amdgcn_tensor_load_to_lds(g0, g1, zero4, zero4, zero8, 0);
}

// ---------------------------------------------------------------------------
// LayerNorm over D_MODEL, output bf16 for WMMA. One block per row (t).
// ---------------------------------------------------------------------------
__global__ void layernorm_bf16_kernel(const float* __restrict__ x,
                                      const float* __restrict__ g,
                                      const float* __restrict__ b,
                                      __bf16* __restrict__ out) {
  const int t   = blockIdx.x;
  const int tid = threadIdx.x;            // 256 threads, 4 elems each
  const float4 v = ((const float4*)(x + (size_t)t * D_MODEL))[tid];
  float s  = v.x + v.y + v.z + v.w;
  float s2 = v.x * v.x + v.y * v.y + v.z * v.z + v.w * v.w;

  __shared__ float rs[256], rq[256];
  rs[tid] = s; rq[tid] = s2;
  __syncthreads();
  for (int o = 128; o > 0; o >>= 1) {
    if (tid < o) { rs[tid] += rs[tid + o]; rq[tid] += rq[tid + o]; }
    __syncthreads();
  }
  const float mu   = rs[0] * (1.0f / D_MODEL);
  const float var  = rq[0] * (1.0f / D_MODEL) - mu * mu;
  const float rstd = rsqrtf(var + 1e-5f);

  const int base = tid * 4;
  const float4 gv = ((const float4*)g)[tid];
  const float4 bv = ((const float4*)b)[tid];
  __bf16* o4 = out + (size_t)t * D_MODEL + base;
  o4[0] = f2bf((v.x - mu) * rstd * gv.x + bv.x);
  o4[1] = f2bf((v.y - mu) * rstd * gv.y + bv.y);
  o4[2] = f2bf((v.z - mu) * rstd * gv.z + bv.z);
  o4[3] = f2bf((v.w - mu) * rstd * gv.w + bv.w);
}

// ---------------------------------------------------------------------------
// fp32 [R][C] -> bf16 transposed [C][R] (so GEMM B-fragments are contiguous).
// ---------------------------------------------------------------------------
__global__ void transpose_cast_bf16(const float* __restrict__ in,
                                    __bf16* __restrict__ out,
                                    int R, int Ccols) {
  size_t idx = (size_t)blockIdx.x * blockDim.x + threadIdx.x;
  size_t total = (size_t)R * Ccols;
  if (idx >= total) return;
  int r = (int)(idx % R);
  int c = (int)(idx / R);
  out[idx] = f2bf(in[(size_t)r * Ccols + c]);   // out[c][r]
}

// ---------------------------------------------------------------------------
// WMMA bf16 GEMM with TDM double buffering:
//   C[M][N] (f32) = A[M][K](bf16) * Bt[N][K]^T (+ addend)
// Block = 128 threads = 4 waves. Block tile = 64M x 64N, K chunk = 64.
// Wave 0 drives the Tensor Data Mover; everyone computes from LDS.
// sched_group_barrier pins the per-chunk schedule: all 20 ds_load_b128
// first (one clause, one partial DScnt wait), then 8 back-to-back WMMAs.
// ---------------------------------------------------------------------------
__global__ void gemm_wmma_bf16_tdm(const __bf16* __restrict__ A,
                                   const __bf16* __restrict__ Bt,
                                   const float*  __restrict__ addend,
                                   float* __restrict__ C,
                                   int M, int N, int K) {
  __shared__ __bf16 Asm[2][64 * LDS_PITCH];
  __shared__ __bf16 Bsm[2][64 * LDS_PITCH];

  const int lane  = threadIdx.x & 31;
  const int wave  = threadIdx.x >> 5;
  const int l15   = lane & 15;
  const int hi    = lane >> 4;                  // lane half: 0 or 1
  const int tileM = blockIdx.y * 64;            // 4 x 16-row sub-tiles
  const int blockN = blockIdx.x * 64;           // 4 waves x 16 cols
  const __bf16* gA = A  + (size_t)tileM  * K;
  const __bf16* gB = Bt + (size_t)blockN * K;
  const int nch = K >> 6;

  if (wave == 0) {
    tdm_load_tile64(gA, lds_off(&Asm[0][0]), (unsigned)K, (unsigned)(M - tileM), (unsigned)K);
    tdm_load_tile64(gB, lds_off(&Bsm[0][0]), (unsigned)K, (unsigned)(N - blockN), (unsigned)K);
    __builtin_amdgcn_s_wait_tensorcnt(0);
  }
  __syncthreads();

  v8f acc[4] = {};

  for (int c = 0; c < nch; ++c) {
    const int buf = c & 1;
    // Kick off DMA for the next chunk into the other buffer while we compute.
    if (wave == 0 && (c + 1) < nch) {
      const int k1 = (c + 1) * TILE_K;
      tdm_load_tile64(gA + k1, lds_off(&Asm[buf ^ 1][0]),
                      (unsigned)(K - k1), (unsigned)(M - tileM), (unsigned)K);
      tdm_load_tile64(gB + k1, lds_off(&Bsm[buf ^ 1][0]),
                      (unsigned)(K - k1), (unsigned)(N - blockN), (unsigned)K);
    }

    const __bf16* as = &Asm[buf][0] + l15 * LDS_PITCH + hi * 8;
    const __bf16* bs = &Bsm[buf][0] + (wave * 16 + l15) * LDS_PITCH + hi * 8;

    // ---- load ALL fragments of this 64-K chunk ----
    v16bf bv[2];
    v16bf av[2][4];
#pragma unroll
    for (int ks2 = 0; ks2 < 2; ++ks2) {
      const int ks = ks2 * 32;
      // B fragment: col n = wave*16+l15, K chunks ks+hi*8..+7 / ks+16+hi*8..+7
      bv[ks2] = ld_frag(bs + ks, bs + ks + 16);
#pragma unroll
      for (int mi = 0; mi < 4; ++mi) {
        // A fragment: row m = mi*16+l15, K chunks ks+hi*8..+7 / ks+16+hi*8..+7
        const __bf16* ap = as + mi * (16 * LDS_PITCH) + ks;
        av[ks2][mi] = ld_frag(ap, ap + 16);
      }
    }
    // ---- then run all 8 WMMAs back-to-back ----
#pragma unroll
    for (int ks2 = 0; ks2 < 2; ++ks2) {
#pragma unroll
      for (int mi = 0; mi < 4; ++mi) {
        acc[mi] = __builtin_amdgcn_wmma_f32_16x16x32_bf16(
            false, av[ks2][mi], false, bv[ks2], (short)0, acc[mi], false, false);
      }
    }
    // Scheduler hint: form the chunk as {20x DS-read} then {8x WMMA} so the
    // matrix ops issue back-to-back behind a single (partial) DScnt wait.
    __builtin_amdgcn_sched_group_barrier(0x100, 20, 0);  // DS read group
    __builtin_amdgcn_sched_group_barrier(0x008, 8, 0);   // MFMA/WMMA group

    if (wave == 0 && (c + 1) < nch) __builtin_amdgcn_s_wait_tensorcnt(0);
    __syncthreads();   // compiler waits DScnt first -> safe to overwrite buf^1
  }

  // D layout: VGPR r -> row tileM + mi*16 + hi*8 + r, col blockN + wave*16 + l15
  const int n = blockN + wave * 16 + l15;
#pragma unroll
  for (int mi = 0; mi < 4; ++mi) {
#pragma unroll
    for (int r = 0; r < 8; ++r) {
      const int m = tileM + mi * 16 + hi * 8 + r;
      const size_t off = (size_t)m * N + n;
      float v = acc[mi][r];
      if (addend) v += addend[off];
      C[off] = v;
    }
  }
}

// ---------------------------------------------------------------------------
// Causal depthwise conv (K=4) + bias + SiLU on x_proj = xz[:, :D_INNER].
// ---------------------------------------------------------------------------
__global__ void conv_silu_kernel(const float* __restrict__ xz,
                                 const float* __restrict__ conv_w,
                                 const float* __restrict__ conv_b,
                                 float* __restrict__ x_conv) {
  const int idx = blockIdx.x * blockDim.x + threadIdx.x; // t*D_INNER + d
  const int d = idx & (D_INNER - 1);
  const int t = idx >> 11;                               // / D_INNER
  float acc = conv_b[d];
  const float* w = conv_w + d * D_CONV;
#pragma unroll
  for (int j = 0; j < D_CONV; ++j) {
    const int tt = t - (D_CONV - 1) + j;
    if (tt >= 0) acc = fmaf(xz[(size_t)tt * NXZ + d], w[j], acc);
  }
  x_conv[idx] = acc / (1.0f + __expf(-acc));             // silu
}

// ---------------------------------------------------------------------------
// ssm = x_conv @ W_x  (N=33, too narrow for WMMA). One block per t,
// 4-way split-K with LDS reduction.
// ---------------------------------------------------------------------------
__global__ void ssm_proj_kernel(const float* __restrict__ x_conv,
                                const float* __restrict__ W_x,
                                float* __restrict__ ssm) {
  const int t   = blockIdx.x;
  const int tid = threadIdx.x;       // 256
  const int seg = tid >> 6;          // 0..3 -> K slice of 512
  const int c   = tid & 63;          // 0..63, <33 active
  __shared__ float part[4][33];
  if (c < 2 * D_STATE + 1) {
    const float* row = x_conv + (size_t)t * D_INNER + seg * 512;
    const float* w   = W_x + (size_t)seg * 512 * (2 * D_STATE + 1) + c;
    float acc = 0.0f;
    for (int k = 0; k < 512; ++k)
      acc = fmaf(row[k], w[(size_t)k * (2 * D_STATE + 1)], acc);
    part[seg][c] = acc;
  }
  __syncthreads();
  if (tid < 2 * D_STATE + 1)
    ssm[t * (2 * D_STATE + 1) + tid] =
        part[0][tid] + part[1][tid] + part[2][tid] + part[3][tid];
}

// ---------------------------------------------------------------------------
// Selective-scan: 1 thread per channel d, 16 states in registers, LDS stages
// the shared 33-float ssm row per timestep. Emits y * silu(z) in bf16.
// ---------------------------------------------------------------------------
__global__ void scan_kernel(const float* __restrict__ ssm,
                            const float* __restrict__ x_conv,
                            const float* __restrict__ xz,
                            const float* __restrict__ W_dt,
                            const float* __restrict__ b_dt,
                            const float* __restrict__ A_log,
                            const float* __restrict__ Dp,
                            __bf16* __restrict__ ybf) {
  const int d  = blockIdx.x * blockDim.x + threadIdx.x;
  const float wd = W_dt[d];
  const float bd = b_dt[d];
  const float Dd = Dp[d];
  float A[D_STATE], h[D_STATE];
#pragma unroll
  for (int n = 0; n < D_STATE; ++n) {
    A[n] = -__expf(A_log[d * D_STATE + n]);
    h[n] = 0.0f;
  }
  __shared__ float srow[2 * D_STATE + 1];

  for (int t = 0; t < TLEN; ++t) {
    if (threadIdx.x < 2 * D_STATE + 1)
      srow[threadIdx.x] = ssm[t * (2 * D_STATE + 1) + threadIdx.x];
    __syncthreads();

    const float pre = fmaf(srow[0], wd, bd);
    const float dt  = (pre > 20.0f) ? pre : log1pf(__expf(pre));   // softplus
    const float xc  = x_conv[(size_t)t * D_INNER + d];
    const float dtx = dt * xc;
    float y = 0.0f;
#pragma unroll
    for (int n = 0; n < D_STATE; ++n) {
      const float ab = __expf(dt * A[n]);
      h[n] = fmaf(ab, h[n], dtx * srow[1 + n]);       // h = A_bar*h + Bx
      y = fmaf(h[n], srow[1 + D_STATE + n], y);       // y += h * C
    }
    y = fmaf(xc, Dd, y);                               // + x_conv * D
    const float zv = xz[(size_t)t * NXZ + D_INNER + d];
    const float yz = y * (zv / (1.0f + __expf(-zv)));  // * silu(z)
    ybf[(size_t)t * D_INNER + d] = f2bf(yz);
    __syncthreads();
  }
}

// ---------------------------------------------------------------------------
extern "C" void kernel_launch(void* const* d_in, const int* in_sizes, int n_in,
                              void* d_out, int out_size, void* d_ws, size_t ws_size,
                              hipStream_t stream) {
  const float* x      = (const float*)d_in[0];
  const float* W_in   = (const float*)d_in[1];
  const float* conv_w = (const float*)d_in[2];
  const float* conv_b = (const float*)d_in[3];
  const float* W_x    = (const float*)d_in[4];
  const float* W_dt   = (const float*)d_in[5];
  const float* b_dt   = (const float*)d_in[6];
  const float* A_log  = (const float*)d_in[7];
  const float* Dp     = (const float*)d_in[8];
  const float* W_out  = (const float*)d_in[9];
  const float* ln_g   = (const float*)d_in[10];
  const float* ln_b   = (const float*)d_in[11];
  float* out = (float*)d_out;

  char* ws = (char*)d_ws;
  size_t off = 0;
  __bf16* Wt_in  = (__bf16*)(ws + off); off += (size_t)NXZ * D_MODEL * 2;      // 8 MB
  __bf16* Wt_out = (__bf16*)(ws + off); off += (size_t)D_MODEL * D_INNER * 2;  // 4 MB
  __bf16* xn     = (__bf16*)(ws + off); off += (size_t)TLEN * D_MODEL * 2;     // 4 MB
  float*  xz     = (float*)(ws + off);  off += (size_t)TLEN * NXZ * 4;         // 32 MB
  float*  xcv    = (float*)(ws + off);  off += (size_t)TLEN * D_INNER * 4;     // 16 MB
  float*  ssm    = (float*)(ws + off);  off += (size_t)TLEN * (2 * D_STATE + 1) * 4;
  __bf16* ybf    = (__bf16*)(ws + off); off += (size_t)TLEN * D_INNER * 2;     // 8 MB

  // 1) LayerNorm -> bf16 activations
  layernorm_bf16_kernel<<<TLEN, 256, 0, stream>>>(x, ln_g, ln_b, xn);
  // 2) Weight precasts (transposed so B fragments are K-contiguous)
  transpose_cast_bf16<<<(D_MODEL * NXZ) / 256, 256, 0, stream>>>(W_in, Wt_in, D_MODEL, NXZ);
  transpose_cast_bf16<<<(D_INNER * D_MODEL) / 256, 256, 0, stream>>>(W_out, Wt_out, D_INNER, D_MODEL);
  // 3) GEMM1: xz = xn @ W_in  (2048x4096, K=1024), TDM + WMMA
  gemm_wmma_bf16_tdm<<<dim3(NXZ / 64, TLEN / 64), 128, 0, stream>>>(
      xn, Wt_in, nullptr, xz, TLEN, NXZ, D_MODEL);
  // 4) depthwise conv + SiLU
  conv_silu_kernel<<<(TLEN * D_INNER) / 256, 256, 0, stream>>>(xz, conv_w, conv_b, xcv);
  // 5) narrow projection to (dt_raw | B | C)
  ssm_proj_kernel<<<TLEN, 256, 0, stream>>>(xcv, W_x, ssm);
  // 6) selective scan -> y * silu(z) in bf16
  scan_kernel<<<D_INNER / 256, 256, 0, stream>>>(ssm, xcv, xz, W_dt, b_dt, A_log, Dp, ybf);
  // 7) GEMM2 + residual: out = ybf @ W_out + x  (2048x1024, K=2048), TDM + WMMA
  gemm_wmma_bf16_tdm<<<dim3(D_MODEL / 64, TLEN / 64), 128, 0, stream>>>(
      ybf, Wt_out, x, out, TLEN, D_MODEL, D_INNER);
}
